// convLR3_44470091383400
// MI455X (gfx1250) — matrix operand order repaired and checked by reading, not compiled, not run on hardware
//
#include <hip/hip_runtime.h>

// ---------------------------------------------------------------------------
// 1x1 conv as GEMM:  out[n,o,hw] = ReLU( sum_c fea[n,c,hw] * W[o,c,0,4] + b[o] )
// M = 512 (o), N = 32768 (n*hw), K = 2048 (c).
// bf16 WMMA (v_wmma_f32_16x16x32_bf16), f32 accumulate.
// Roofline: 68.7 GFLOP vs 340 MB -> ~14.6us mem @ 23.3TB/s, ~balanced w/ bf16 math.
// Grid is linearized with the o-tile fastest so the 4 o-blocks sharing an fea
// s-tile are adjacent in dispatch order: fea is read from HBM once, reused via
// the 192MB L2 by the other 3 blocks (keeps HBM traffic at ~268MB, not 4x).
// ---------------------------------------------------------------------------

typedef __attribute__((ext_vector_type(16))) __bf16       v16bf;
typedef __attribute__((ext_vector_type(8)))  float        v8f;
typedef __attribute__((ext_vector_type(8)))  unsigned int v8u;

union FragU {
  v8u   u;
  v16bf bf;
};

// fp32 pair -> packed bf16 word via v_perm_b32.
// Round-half-up (+0x8000) then pick bytes [b3,b2 | a3,a2].
__device__ __forceinline__ unsigned int pack_bf16(float a, float b) {
  unsigned int ua = __builtin_bit_cast(unsigned int, a) + 0x8000u;
  unsigned int ub = __builtin_bit_cast(unsigned int, b) + 0x8000u;
  return __builtin_amdgcn_perm(ub, ua, 0x07060302u);
}

// ---------------------------------------------------------------------------
// Kernel 0: gather W[:, :, 0, 4] (stride-9 fp32) into bf16 A-fragments laid out
// exactly as V_WMMA_F32_16X16X32_BF16 wants its A matrix (16x32, MxK):
//   lane half h = L/16, row m = L%16,
//   element e: K = 8h + (e%8) + 16*(e/8)
// Frag (mt, kt) is 32 lanes * 32B contiguous -> coalesced v8u (2x b128) loads.
// ---------------------------------------------------------------------------
__global__ __launch_bounds__(256) void prepack_w_kernel(
    const float* __restrict__ W, unsigned short* __restrict__ Apack) {
  int idx = blockIdx.x * 256 + threadIdx.x;   // 512*2048 entries
  if (idx >= 512 * 2048) return;
  int o = idx >> 11;
  int c = idx & 2047;
  float w = W[(size_t)o * (2048 * 9) + (size_t)c * 9 + 4];
  unsigned int uw = __builtin_bit_cast(unsigned int, w) + 0x8000u;

  int mt = o >> 4, r  = o & 15;
  int kt = c >> 5, kk = c & 31;
  int a   = kk >> 4;        // e/8
  int rem = kk & 15;
  int h   = rem >> 3;       // lane half
  int c3  = rem & 7;        // e%8
  int lane = h * 16 + r;
  int e    = a * 8 + c3;
  Apack[((((size_t)mt * 64 + kt) * 32 + lane) << 4) + e] = (unsigned short)(uw >> 16);
}

// ---------------------------------------------------------------------------
// Main GEMM kernel.
// Block: 256 threads (8 waves). Block tile: 128 (o) x 128 (s). Wave tile: 32x64.
// K loop: 64 steps of 32, fea tile staged through double-buffered LDS as
// K-pair-packed bf16 in an [s][k2] layout (stride 20 words for bank spread).
// B fragment (32x16 KxN): lane n = L%16, K = 8*(L/16) + e%8 + 16*(e/8)  -> 32B/lane from LDS.
// ---------------------------------------------------------------------------
#define LDS_STRIDE 20

__global__ __launch_bounds__(256) void conv1x1_wmma_kernel(
    const float* __restrict__ fea, const unsigned short* __restrict__ Apack,
    const float* __restrict__ bias, float* __restrict__ out) {
  __shared__ unsigned int Bt0[128 * LDS_STRIDE];
  __shared__ unsigned int Bt1[128 * LDS_STRIDE];

  const int tid    = threadIdx.x;
  const int lane   = tid & 31;
  const int wid    = tid >> 5;
  const int wave_n = wid & 1;   // 2 waves across s (64 each)
  const int wave_m = wid >> 1;  // 4 waves across o (32 each)

  // Linearized grid, o-tile fastest: bid = s_tile*4 + o_tile.
  const int bid    = blockIdx.x;
  const int o0     = (bid & 3) * 128;        // 4 tiles of 128 over oc=512
  const int s_tile = bid >> 2;               // 256 s-tiles
  const int img    = s_tile >> 5;            // 8 images
  const int s0     = (s_tile & 31) * 128;    // 32 tiles of 128 over hw=4096

  const float* __restrict__ feaN   = fea + (size_t)img * 2048 * 4096;
  const v8u*   __restrict__ apackv = (const v8u*)Apack;

  // staging decomposition: each thread owns 2 items of a 16(k2) x 32(s4) grid
  const int k2a = tid >> 5;        // 0..7
  const int k2b = 8 + (tid >> 5);  // 8..15
  const int s4  = (tid & 31) << 2; // s offset 0..124 step 4

  v8f acc[2][4];
#pragma unroll
  for (int i = 0; i < 2; ++i)
#pragma unroll
    for (int j = 0; j < 4; ++j) acc[i][j] = (v8f){0.f, 0.f, 0.f, 0.f, 0.f, 0.f, 0.f, 0.f};

  float4 xA, yA, xB, yB;

  auto fetch = [&](int kt) {
    const size_t base = (size_t)(kt * 32) * 4096 + (size_t)(s0 + s4);
    xA = *(const float4*)(feaN + base + (size_t)(2 * k2a    ) * 4096);
    yA = *(const float4*)(feaN + base + (size_t)(2 * k2a + 1) * 4096);
    xB = *(const float4*)(feaN + base + (size_t)(2 * k2b    ) * 4096);
    yB = *(const float4*)(feaN + base + (size_t)(2 * k2b + 1) * 4096);
  };

  auto commit = [&](unsigned int* B) {
    B[(s4 + 0) * LDS_STRIDE + k2a] = pack_bf16(xA.x, yA.x);
    B[(s4 + 1) * LDS_STRIDE + k2a] = pack_bf16(xA.y, yA.y);
    B[(s4 + 2) * LDS_STRIDE + k2a] = pack_bf16(xA.z, yA.z);
    B[(s4 + 3) * LDS_STRIDE + k2a] = pack_bf16(xA.w, yA.w);
    B[(s4 + 0) * LDS_STRIDE + k2b] = pack_bf16(xB.x, yB.x);
    B[(s4 + 1) * LDS_STRIDE + k2b] = pack_bf16(xB.y, yB.y);
    B[(s4 + 2) * LDS_STRIDE + k2b] = pack_bf16(xB.z, yB.z);
    B[(s4 + 3) * LDS_STRIDE + k2b] = pack_bf16(xB.w, yB.w);
  };

  const int sn = lane & 15;
  const int h  = lane >> 4;

  auto compute = [&](const unsigned int* Bb, int kt) {
    FragU a0, a1;
    const int mt0 = (o0 >> 4) + wave_m * 2;
    a0.u = apackv[((size_t)(mt0    ) * 64 + kt) * 32 + lane];
    a1.u = apackv[((size_t)(mt0 + 1) * 64 + kt) * 32 + lane];
#pragma unroll
    for (int nf = 0; nf < 4; ++nf) {
      const unsigned int* p = Bb + (wave_n * 64 + nf * 16 + sn) * LDS_STRIDE + 4 * h;
      uint4 lo = *(const uint4*)(p);
      uint4 hi = *(const uint4*)(p + 8);
      FragU bfr;
      bfr.u = (v8u){lo.x, lo.y, lo.z, lo.w, hi.x, hi.y, hi.z, hi.w};
      acc[0][nf] = __builtin_amdgcn_wmma_f32_16x16x32_bf16(
          false, a0.bf, false, bfr.bf, (short)0, acc[0][nf], false, false);
      acc[1][nf] = __builtin_amdgcn_wmma_f32_16x16x32_bf16(
          false, a1.bf, false, bfr.bf, (short)0, acc[1][nf], false, false);
    }
  };

  fetch(0);
  commit(Bt0);
  __syncthreads();

  // Explicit double-buffer unroll: buffer selection is compile-time constant.
  for (int kt = 0; kt < 64; kt += 2) {
    // even step: compute from Bt0, stage kt+1 into Bt1
    if (kt + 1 < 64) fetch(kt + 1);
    compute(Bt0, kt);
    if (kt + 1 < 64) commit(Bt1);
    __syncthreads();
    if (kt + 1 >= 64) break;
    // odd step: compute from Bt1, stage kt+2 into Bt0
    if (kt + 2 < 64) fetch(kt + 2);
    compute(Bt1, kt + 1);
    if (kt + 2 < 64) commit(Bt0);
    __syncthreads();
  }

  // Epilogue: D layout 16x16 f32 -> VGPR r holds rows {r, r+8}, lane = column.
#pragma unroll
  for (int mf = 0; mf < 2; ++mf) {
#pragma unroll
    for (int r = 0; r < 8; ++r) {
      const int o  = o0 + wave_m * 32 + mf * 16 + h * 8 + r;
      const float bv = bias[o];
#pragma unroll
      for (int nf = 0; nf < 4; ++nf) {
        const int s = s0 + wave_n * 64 + nf * 16 + sn;
        float v = acc[mf][nf][r] + bv;
        out[((size_t)img * 512 + o) * 4096 + s] = v > 0.f ? v : 0.f;
      }
    }
  }
}

extern "C" void kernel_launch(void* const* d_in, const int* in_sizes, int n_in,
                              void* d_out, int out_size, void* d_ws, size_t ws_size,
                              hipStream_t stream) {
  const float* fea = (const float*)d_in[0];
  const float* W   = (const float*)d_in[1];
  const float* b   = (const float*)d_in[2];
  float* out = (float*)d_out;

  unsigned short* Apack = (unsigned short*)d_ws;  // 512*2048*2 = 2 MB scratch

  prepack_w_kernel<<<(512 * 2048 + 255) / 256, 256, 0, stream>>>(W, Apack);

  // 1024 blocks: bid = s_tile*4 + o_tile (o fastest for L2 reuse of fea tiles)
  conv1x1_wmma_kernel<<<1024, 256, 0, stream>>>(fea, Apack, b, out);
}
